// NeighborhoodAttention_47665547051100
// MI455X (gfx1250) — matrix operand order repaired and checked by reading, not compiled, not run on hardware
//
#include <hip/hip_runtime.h>
#include <hip/hip_bf16.h>
#include <math.h>

// ---------------------------------------------------------------------------
// Problem constants (reference: DIM=512, HEAD_DIM=64, NH=8, KERNEL=7, H=W=64)
// ---------------------------------------------------------------------------
#define DIM      512
#define HEAD_DIM 64
#define NH       8
#define HH       64
#define WW       64
#define NPIX     4096        // H*W
#define KWIN     7
#define KK       49          // 7*7

typedef _Float16 v4h  __attribute__((ext_vector_type(4)));
typedef _Float16 v8h  __attribute__((ext_vector_type(8)));
typedef _Float16 v16h __attribute__((ext_vector_type(16)));
typedef float    v8f  __attribute__((ext_vector_type(8)));

// ---------------------------------------------------------------------------
// Kernel 1 & 5: GEMM  C[M,N] = A[M,K] * B[K,N] (+ bias[row])
// f32 in/out, f16 WMMA compute, f32 accumulate.
// Block: 256 threads (8 waves), 64x64 output tile, BK=64 staged via LDS.
// Each wave: 16x32 strip -> two 16x16 f32 accumulators, 4 WMMAs / K-block.
// ---------------------------------------------------------------------------
#define LDT 72   // LDS row pitch in f16 (64 + 8 pad); 144B = multiple of 16B

__global__ __launch_bounds__(256) void na_gemm_f16wmma(
    const float* __restrict__ A, const float* __restrict__ B,
    float* __restrict__ C, const float* __restrict__ bias,
    int M, int N, int K)
{
    __shared__ _Float16 As[64 * LDT];  // As[m][k]
    __shared__ _Float16 Bs[64 * LDT];  // Bs[n][k]  (B transposed in LDS)

    const int tid  = threadIdx.x;
    const int lane = tid & 31;
    const int wv   = tid >> 5;
    const int half = lane >> 4;     // wave32: lanes 0-15 vs 16-31
    const int l16  = lane & 15;
    const int wm   = wv & 3;        // 4 M-subtiles of 16
    const int wn   = wv >> 2;       // 2 N-subtiles of 32

    const int nblk = N >> 6;
    const int bm   = blockIdx.x / nblk;
    const int bn   = blockIdx.x % nblk;

    v8f acc0 = {0.f,0.f,0.f,0.f,0.f,0.f,0.f,0.f};
    v8f acc1 = {0.f,0.f,0.f,0.f,0.f,0.f,0.f,0.f};

    for (int kb = 0; kb < K; kb += 64) {
        // Prefetch hint for next K-block (global_prefetch_b8)
        if (kb + 64 < K) {
            __builtin_prefetch(A + (size_t)(bm*64 + (tid >> 4))*K + kb + 64 + ((tid & 15) << 2), 0, 1);
            __builtin_prefetch(B + (size_t)(kb + 64 + (tid >> 4))*N + bn*64 + ((tid & 15) << 2), 0, 1);
        }

        // Stage A tile (64x64 f32 -> f16), row-major, coalesced float4 loads.
        #pragma unroll
        for (int i = 0; i < 4; ++i) {
            int f   = tid + i * 256;          // float4 id: 0..1023
            int row = f >> 4;                 // 0..63 (m local)
            int c4  = (f & 15) << 2;          // 0..60 (k local)
            float4 x = *(const float4*)(A + (size_t)(bm*64 + row)*K + kb + c4);
            v4h hx = { (_Float16)x.x, (_Float16)x.y, (_Float16)x.z, (_Float16)x.w };
            *(v4h*)&As[row * LDT + c4] = hx;
        }
        // Stage B tile (64x64 f32 -> f16), transposed into Bs[n][k].
        #pragma unroll
        for (int i = 0; i < 4; ++i) {
            int f   = tid + i * 256;
            int row = f >> 4;                 // k local
            int c4  = (f & 15) << 2;          // n local
            float4 x = *(const float4*)(B + (size_t)(kb + row)*N + bn*64 + c4);
            Bs[(c4 + 0) * LDT + row] = (_Float16)x.x;
            Bs[(c4 + 1) * LDT + row] = (_Float16)x.y;
            Bs[(c4 + 2) * LDT + row] = (_Float16)x.z;
            Bs[(c4 + 3) * LDT + row] = (_Float16)x.w;
        }
        __syncthreads();

        #pragma unroll
        for (int k0 = 0; k0 < 64; k0 += 32) {
            // A fragment: contiguous k order (k = 16*half + i), same as B.
            // lanes 0-15: K = k0 + 0..15; lanes 16-31: K = k0 + 16..31
            const _Float16* ap = &As[(wm*16 + l16) * LDT + k0 + half*16];
            v8h alo = *(const v8h*)ap;
            v8h ahi = *(const v8h*)(ap + 8);
            v16h a;
            #pragma unroll
            for (int i = 0; i < 8; ++i) { a[i] = alo[i]; a[i+8] = ahi[i]; }

            // B fragment (tile 0): lanes 0-15: K = k0+0..15; lanes 16-31: K = k0+16..31
            const _Float16* bp0 = &Bs[(wn*32 + l16) * LDT + k0 + half*16];
            v8h b0lo = *(const v8h*)bp0;
            v8h b0hi = *(const v8h*)(bp0 + 8);
            v16h b0;
            #pragma unroll
            for (int i = 0; i < 8; ++i) { b0[i] = b0lo[i]; b0[i+8] = b0hi[i]; }
            acc0 = __builtin_amdgcn_wmma_f32_16x16x32_f16(false, a, false, b0,
                                                          (short)0, acc0, false, false);

            // B fragment (tile 1)
            const _Float16* bp1 = &Bs[(wn*32 + 16 + l16) * LDT + k0 + half*16];
            v8h b1lo = *(const v8h*)bp1;
            v8h b1hi = *(const v8h*)(bp1 + 8);
            v16h b1;
            #pragma unroll
            for (int i = 0; i < 8; ++i) { b1[i] = b1lo[i]; b1[i+8] = b1hi[i]; }
            acc1 = __builtin_amdgcn_wmma_f32_16x16x32_f16(false, a, false, b1,
                                                          (short)0, acc1, false, false);
        }
        __syncthreads();
    }

    // Epilogue: C VGPR v holds M=v (lanes 0-15) / M=v+8 (lanes 16-31), N=lane&15
    const int col0 = bn*64 + wn*32 + l16;
    #pragma unroll
    for (int v = 0; v < 8; ++v) {
        int row = bm*64 + wm*16 + v + half*8;
        float bv = bias ? bias[row] : 0.0f;
        C[(size_t)row * N + col0]      = acc0[v] + bv;
        C[(size_t)row * N + col0 + 16] = acc1[v] + bv;
    }
}

// ---------------------------------------------------------------------------
// Kernel 2: depthwise 3x3 conv (cross-correlation, SAME zero pad) + bias
// one thread per (channel, pixel); 1536*4096 threads
// ---------------------------------------------------------------------------
__global__ __launch_bounds__(256) void na_dwconv(
    const float* __restrict__ in, const float* __restrict__ wdw,
    const float* __restrict__ bdw, float* __restrict__ out)
{
    int tid = blockIdx.x * 256 + threadIdx.x;   // 0 .. 1536*4096-1
    int c   = tid >> 12;
    int pix = tid & 4095;
    int h = pix >> 6, w = pix & 63;

    const float* ip = in + ((size_t)c << 12);
    const float* wp = wdw + c * 9;
    float s = bdw[c];
    #pragma unroll
    for (int dh = -1; dh <= 1; ++dh) {
        int hh = h + dh;
        if ((unsigned)hh >= 64u) continue;
        #pragma unroll
        for (int dw = -1; dw <= 1; ++dw) {
            int ww2 = w + dw;
            if ((unsigned)ww2 >= 64u) continue;
            s += ip[(hh << 6) + ww2] * wp[(dh + 1) * 3 + (dw + 1)];
        }
    }
    out[tid] = s;
}

// ---------------------------------------------------------------------------
// Kernel 3: split q/k/v + RoPE (first 32 of 64 head dims rotated, d_theta=16)
// thread id encodes (part[3], head[8], pixel[4096], e[64]); writes
// q/k/v as [head][pixel][64] f32
// ---------------------------------------------------------------------------
__global__ __launch_bounds__(256) void na_rope_split(
    const float* __restrict__ conv, const float* __restrict__ freqs,
    float* __restrict__ q, float* __restrict__ k, float* __restrict__ v)
{
    int tid  = blockIdx.x * 256 + threadIdx.x;  // 0 .. 3*512*4096-1
    int e    = tid & 63;
    int pix  = (tid >> 6) & 4095;
    int head = (tid >> 18) & 7;
    int part = tid >> 21;                       // 0=q, 1=k, 2=v

    int h = pix >> 6, w = pix & 63;
    int c = part * DIM + head * HEAD_DIM + e;
    float val = conv[(size_t)c * NPIX + pix];

    if (part < 2 && e < 32) {
        int i = e & 15;
        float p = (i < 8) ? ((2.0f * h) / 63.0f - 1.0f)    // pos_y
                          : ((2.0f * w) / 63.0f - 1.0f);   // pos_x
        float fr = freqs[head * 8 + (i & 7)];
        float th = p * fr;
        float cs = __cosf(th), sn = __sinf(th);
        if (e < 16) {
            float x2 = conv[(size_t)(c + 16) * NPIX + pix];
            val = val * cs - x2 * sn;
        } else {
            float x1 = conv[(size_t)(c - 16) * NPIX + pix];
            val = val * cs + x1 * sn;
        }
    }
    float* dst = (part == 0) ? q : (part == 1) ? k : v;
    dst[(((size_t)(head << 12) + pix) << 6) + e] = val;
}

// ---------------------------------------------------------------------------
// Kernel 4: 7x7 neighborhood attention. One wave32 per (head, pixel).
// Lanes cover 49 key slots (2/lane) for QK + softmax (shfl_xor reductions),
// then cover the 64 output dims (2/lane) for A*V. Output written
// channel-major [head*64+e][pixel] so it feeds the output GEMM directly.
// ---------------------------------------------------------------------------
__global__ __launch_bounds__(256) void na_na2d(
    const float* __restrict__ q, const float* __restrict__ k,
    const float* __restrict__ v, float* __restrict__ out)
{
    __shared__ float attn_s[8][64];

    const int lane = threadIdx.x & 31;
    const int wv   = threadIdx.x >> 5;
    const int gw   = blockIdx.x * 8 + wv;      // 0 .. 8*4096-1
    const int head = gw >> 12;
    const int pix  = gw & 4095;
    const int h = pix >> 6, w = pix & 63;

    int si = h - 3; si = si < 0 ? 0 : (si > 57 ? 57 : si);
    int sj = w - 3; sj = sj < 0 ? 0 : (sj > 57 ? 57 : sj);

    const float* qp = q + (((size_t)(head << 12) + pix) << 6);
    const float* kh = k + ((size_t)head << 18);

    // logits (scale = 64^-0.5 = 0.125)
    float lg[2];
    #pragma unroll
    for (int s2 = 0; s2 < 2; ++s2) {
        int p = lane + 32 * s2;
        float d = -1e30f;
        if (p < KK) {
            int kr = si + p / 7, kc = sj + p % 7;
            const float* kp = kh + (((size_t)(kr << 6) + kc) << 6);
            float acc = 0.f;
            #pragma unroll
            for (int e = 0; e < 64; e += 4) {
                float4 qv = *(const float4*)(qp + e);
                float4 kv = *(const float4*)(kp + e);
                acc += qv.x*kv.x + qv.y*kv.y + qv.z*kv.z + qv.w*kv.w;
            }
            d = acc * 0.125f;
        }
        lg[s2] = d;
    }

    // softmax across the wave (wave32 reductions)
    float m = fmaxf(lg[0], lg[1]);
    #pragma unroll
    for (int off = 16; off > 0; off >>= 1) m = fmaxf(m, __shfl_xor(m, off, 32));
    float e0 = __expf(lg[0] - m);
    float e1 = (lane < 17) ? __expf(lg[1] - m) : 0.f;
    float s = e0 + e1;
    #pragma unroll
    for (int off = 16; off > 0; off >>= 1) s += __shfl_xor(s, off, 32);
    float inv = 1.0f / s;

    attn_s[wv][lane] = e0 * inv;
    if (lane < 17) attn_s[wv][lane + 32] = e1 * inv;
    __syncthreads();

    // A*V: lanes now cover output dims e = lane, lane+32
    const float* vh = v + ((size_t)head << 18);
    float o0 = 0.f, o1 = 0.f;
    #pragma unroll 7
    for (int p = 0; p < KK; ++p) {
        int kr = si + p / 7, kc = sj + p % 7;
        const float* vp = vh + (((size_t)(kr << 6) + kc) << 6);
        float a = attn_s[wv][p];
        o0 += a * vp[lane];
        o1 += a * vp[lane + 32];
    }
    out[(size_t)((head << 6) + lane)      * NPIX + pix] = o0;
    out[(size_t)((head << 6) + lane + 32) * NPIX + pix] = o1;
}

// ---------------------------------------------------------------------------
// Host side
// ---------------------------------------------------------------------------
extern "C" void kernel_launch(void* const* d_in, const int* in_sizes, int n_in,
                              void* d_out, int out_size, void* d_ws, size_t ws_size,
                              hipStream_t stream)
{
    const float* hs    = (const float*)d_in[0];  // (1,512,64,64)
    // d_in[1] = pos (recomputed analytically on device)
    const float* w_qkv = (const float*)d_in[2];  // (1536,512)
    const float* w_dw  = (const float*)d_in[3];  // (1536,1,3,3)
    const float* b_dw  = (const float*)d_in[4];  // (1536,)
    const float* freqs = (const float*)d_in[5];  // (8,8) = [head][i]
    const float* w_out = (const float*)d_in[6];  // (512,512)
    const float* b_out = (const float*)d_in[7];  // (512,)
    float* out = (float*)d_out;                  // (1,512,64,64)

    // workspace layout (floats)
    float* ws       = (float*)d_ws;
    float* qkv_lin  = ws;                                  // 1536*4096
    float* qkv_conv = qkv_lin  + (size_t)3*DIM*NPIX;       // 1536*4096
    float* qb       = qkv_conv + (size_t)3*DIM*NPIX;       // 8*4096*64
    float* kb       = qb + (size_t)NH*NPIX*HEAD_DIM;
    float* vb       = kb + (size_t)NH*NPIX*HEAD_DIM;
    float* attnout  = qkv_lin;                             // reuse (no longer needed)

    // 1) qkv = w_qkv @ hidden : M=1536, N=4096, K=512
    na_gemm_f16wmma<<<(3*DIM/64)*(NPIX/64), 256, 0, stream>>>(
        w_qkv, hs, qkv_lin, nullptr, 3*DIM, NPIX, DIM);

    // 2) depthwise 3x3 conv + bias
    na_dwconv<<<(3*DIM*NPIX)/256, 256, 0, stream>>>(qkv_lin, w_dw, b_dw, qkv_conv);

    // 3) split + RoPE -> q, k, v as [head][pixel][64]
    na_rope_split<<<(3*DIM*NPIX)/256, 256, 0, stream>>>(qkv_conv, freqs, qb, kb, vb);

    // 4) 7x7 neighborhood attention -> channel-major [512][4096]
    na_na2d<<<NPIX, 256, 0, stream>>>(qb, kb, vb, attnout);

    // 5) out = w_out @ attnout + b_out : M=512, N=4096, K=512
    na_gemm_f16wmma<<<(DIM/64)*(NPIX/64), 256, 0, stream>>>(
        w_out, attnout, out, b_out, DIM, NPIX, DIM);
}
